// LrUpsampling_48000554500591
// MI455X (gfx1250) — matrix operand order repaired and checked by reading, not compiled, not run on hardware
//
#include <hip/hip_runtime.h>
#include <math.h>

// ---------------------------------------------------------------------------
// MI455X (gfx1250): 2x TransformerConv (dense attention) + GraphNorm +
// row-normalized Gram output.  All GEMM work on v_wmma_f32_16x16x32_bf16.
// Round-4: hoisted all loop-invariant fast/slow tile-load selection out of
// the k-loops.  Main GEMM loops are branch-free: global_load_b128 ->
// ds_store_b128 -> barrier -> ds_load_b128 pairs -> 4x v_wmma.  Guarded
// scalar paths only run in a separate K-remainder tail loop.
// ---------------------------------------------------------------------------

typedef __attribute__((ext_vector_type(16))) __bf16 v16bf;
typedef __attribute__((ext_vector_type(8)))  __bf16 v8bf;
typedef __attribute__((ext_vector_type(8)))  float  v8f;

#define NN    4096   // nodes
#define LRD   512    // input feature dim
#define DD1   200    // layer-1 feature dim
#define HRD   2048   // layer-2 feature dim
#define NHEAD 4
#define CH1   50     // per-head dim, layer 1
#define CH2   512    // per-head dim, layer 2
#define LD1P  256    // padded ld for layer-1 features

static __device__ __forceinline__ v8f wmma_bf16(v16bf a, v16bf b, v8f c) {
  return __builtin_amdgcn_wmma_f32_16x16x32_bf16(false, a, false, b, (short)0, c,
                                                 false, false);
}

// Build a 16-element bf16 fragment from two aligned contiguous 8-element
// chunks in LDS -> two ds_load_b128.
static __device__ __forceinline__ v16bf frag16(const __bf16* p0, const __bf16* p1) {
  v8bf lo = *(const v8bf*)p0;
  v8bf hi = *(const v8bf*)p1;
  return __builtin_shufflevector(lo, hi, 0, 1, 2, 3, 4, 5, 6, 7,
                                 8, 9, 10, 11, 12, 13, 14, 15);
}

static __device__ __forceinline__ v8bf cvt8(float4 a, float4 b) {
  v8bf o;
  o[0] = (__bf16)a.x; o[1] = (__bf16)a.y; o[2] = (__bf16)a.z; o[3] = (__bf16)a.w;
  o[4] = (__bf16)b.x; o[5] = (__bf16)b.y; o[6] = (__bf16)b.z; o[7] = (__bf16)b.w;
  return o;
}

// Per-k-tile fragment gather + 4x WMMA (wave tile 32x32).
static __device__ __forceinline__ void tile_mma(
    const __bf16 (*As)[40], const __bf16 (*Bs)[40],
    int lane, int wm, int wn, v8f acc[2][2])
{
  const int ml  = lane & 15;
  const int kbA = (lane < 16) ? 0 : 8;
  const int kbB = (lane < 16) ? 0 : 8;
  v16bf af[2], bfr[2];
#pragma unroll
  for (int t = 0; t < 2; ++t) {
    const int row = wm * 32 + t * 16 + ml;
    const int col = wn * 32 + t * 16 + ml;
    af[t]  = frag16(&As[row][kbA], &As[row][kbA + 16]);   // ISA A layout
    bfr[t] = frag16(&Bs[col][kbB], &Bs[col][kbB + 16]);   // ISA B layout
  }
#pragma unroll
  for (int i = 0; i < 2; ++i)
#pragma unroll
    for (int j = 0; j < 2; ++j)
      acc[i][j] = wmma_bf16(af[i], bfr[j], acc[i][j]);
}

// ---------------------------------------------------------------------------
// transpose_f32_to_bf16: Y[m, k] = bf16(X[k, m]).  X is [K, M] row-major.
// ---------------------------------------------------------------------------
__global__ __launch_bounds__(256) void transpose_f32_to_bf16(
    const float* __restrict__ X, __bf16* __restrict__ Y, int K, int M)
{
  __shared__ float t[32][33];
  const int tx = threadIdx.x & 31;
  const int ty = threadIdx.x >> 5;   // 0..7
  const int m0 = blockIdx.x * 32;
  const int k0 = blockIdx.y * 32;
#pragma unroll
  for (int i = 0; i < 4; ++i) {
    const int k = k0 + ty + i * 8;
    t[ty + i * 8][tx] = X[(long)k * M + m0 + tx];
  }
  __syncthreads();
#pragma unroll
  for (int i = 0; i < 4; ++i) {
    const int m = m0 + ty + i * 8;
    Y[(long)m * K + k0 + tx] = (__bf16)t[tx][ty + i * 8];
  }
}

// ---------------------------------------------------------------------------
// gemm_lin: out[m,n] = sum_k A[m,k]*B[k*Nc+n] + bias[n]
//   A bf16 row-major [M, lda]; B fp32 row-major [K, Nc].
//   Writes fp32 and/or bf16 output with ld=ldo; columns >= Nc written as 0.
// Block: 256 thr (8 waves), tile 128x64, wave tile 32x32 (4 WMMA accums).
// ---------------------------------------------------------------------------
__global__ __launch_bounds__(256) void gemm_lin(
    const __bf16* __restrict__ A, int lda,
    const float* __restrict__ B, const float* __restrict__ bias,
    float* __restrict__ outF, __bf16* __restrict__ outB, int ldo,
    int K, int Nc)
{
  __shared__ __align__(16) __bf16 As[128][40];   // [m][k]
  __shared__ __align__(16) __bf16 Bs[64][40];    // [n][k]
  const int tid  = threadIdx.x;
  const int lane = tid & 31;
  const int wave = tid >> 5;
  const int wm = wave >> 1, wn = wave & 1;
  const int bm = blockIdx.x * 128;
  const int bn = blockIdx.y * 64;

  v8f acc[2][2];
#pragma unroll
  for (int i = 0; i < 2; ++i)
#pragma unroll
    for (int j = 0; j < 2; ++j)
#pragma unroll
      for (int r = 0; r < 8; ++r) acc[i][j][r] = 0.0f;

  // Loop-invariant per-thread load coordinates.
  const int ar  = tid >> 1;
  const int ac0 = (tid & 1) << 4;
  const __bf16* ap = A + (long)(bm + ar) * lda;
  const int br  = tid >> 3;           // k row within tile
  const int bc0 = (tid & 7) << 3;     // n col base

  const int nkt   = (K + 31) >> 5;
  const int nFast = K >> 5;           // full 32-wide k tiles
  const bool bnFull = (bn + 64 <= Nc);

  int kt = 0;
  if (bnFull) {
    for (; kt < nFast; ++kt) {        // branch-free main loop
      const int k0 = kt << 5;
      *(v8bf*)&As[ar][ac0]     = *(const v8bf*)(ap + k0 + ac0);
      *(v8bf*)&As[ar][ac0 + 8] = *(const v8bf*)(ap + k0 + ac0 + 8);
      const float4* p = (const float4*)(B + (long)(k0 + br) * Nc + bn + bc0);
      float4 q0 = p[0], q1 = p[1];
      Bs[bc0 + 0][br] = (__bf16)q0.x; Bs[bc0 + 1][br] = (__bf16)q0.y;
      Bs[bc0 + 2][br] = (__bf16)q0.z; Bs[bc0 + 3][br] = (__bf16)q0.w;
      Bs[bc0 + 4][br] = (__bf16)q1.x; Bs[bc0 + 5][br] = (__bf16)q1.y;
      Bs[bc0 + 6][br] = (__bf16)q1.z; Bs[bc0 + 7][br] = (__bf16)q1.w;
      __syncthreads();
      tile_mma(As, Bs, lane, wm, wn, acc);
      __syncthreads();
    }
  } else {
    for (; kt < nFast; ++kt) {        // n-guarded B, full k
      const int k0 = kt << 5;
      *(v8bf*)&As[ar][ac0]     = *(const v8bf*)(ap + k0 + ac0);
      *(v8bf*)&As[ar][ac0 + 8] = *(const v8bf*)(ap + k0 + ac0 + 8);
      const float* bp = B + (long)(k0 + br) * Nc;
#pragma unroll
      for (int j = 0; j < 8; ++j) {
        const int n = bn + bc0 + j;
        Bs[bc0 + j][br] = (n < Nc) ? (__bf16)bp[n] : (__bf16)0.0f;
      }
      __syncthreads();
      tile_mma(As, Bs, lane, wm, wn, acc);
      __syncthreads();
    }
  }
  for (; kt < nkt; ++kt) {            // K remainder: fully guarded
    const int k0 = kt << 5;
    v8bf o0, o1;
#pragma unroll
    for (int j = 0; j < 8; ++j) {
      const int k = k0 + ac0 + j;
      o0[j] = (k < K) ? ap[k] : (__bf16)0.0f;
    }
#pragma unroll
    for (int j = 0; j < 8; ++j) {
      const int k = k0 + ac0 + 8 + j;
      o1[j] = (k < K) ? ap[k] : (__bf16)0.0f;
    }
    *(v8bf*)&As[ar][ac0]     = o0;
    *(v8bf*)&As[ar][ac0 + 8] = o1;
    const int k = k0 + br;
#pragma unroll
    for (int j = 0; j < 8; ++j) {
      const int n = bn + bc0 + j;
      Bs[bc0 + j][br] = (k < K && n < Nc) ? (__bf16)B[(long)k * Nc + n]
                                          : (__bf16)0.0f;
    }
    __syncthreads();
    tile_mma(As, Bs, lane, wm, wn, acc);
    __syncthreads();
  }

  const int nl   = lane & 15;
  const int mofs = (lane < 16) ? 0 : 8;
#pragma unroll
  for (int i = 0; i < 2; ++i)
#pragma unroll
    for (int j = 0; j < 2; ++j) {
      const int gn = bn + wn * 32 + j * 16 + nl;
      const bool valid = (gn < Nc);
      const float bv = valid ? bias[gn] : 0.0f;
#pragma unroll
      for (int r = 0; r < 8; ++r) {
        const int gm = bm + wm * 32 + i * 16 + mofs + r;
        const float v = valid ? (acc[i][j][r] + bv) : 0.0f;
        if (outF) outF[(long)gm * ldo + gn] = v;
        if (outB) outB[(long)gm * ldo + gn] = (__bf16)v;
      }
    }
}

// ---------------------------------------------------------------------------
// gemm_gram: out[m,n] = act(scale * sum_k A[m,ao+k] * B[n,bo+k]), bf16 inputs.
// Used for per-head Q@K^T scores and the final relu(G @ G^T).
// ---------------------------------------------------------------------------
__global__ __launch_bounds__(256) void gemm_gram(
    const __bf16* __restrict__ A, int lda, int ao,
    const __bf16* __restrict__ Bm, int ldb, int bo,
    float* __restrict__ out, int ldo, int K, float scale, int relu)
{
  __shared__ __align__(16) __bf16 As[128][40];
  __shared__ __align__(16) __bf16 Bs[64][40];   // [n][k]
  const int tid  = threadIdx.x;
  const int lane = tid & 31;
  const int wave = tid >> 5;
  const int wm = wave >> 1, wn = wave & 1;
  const int bm = blockIdx.x * 128;
  const int bn = blockIdx.y * 64;

  v8f acc[2][2];
#pragma unroll
  for (int i = 0; i < 2; ++i)
#pragma unroll
    for (int j = 0; j < 2; ++j)
#pragma unroll
      for (int r = 0; r < 8; ++r) acc[i][j][r] = 0.0f;

  const int ar  = tid >> 1;
  const int ac0 = (tid & 1) << 4;
  const __bf16* ap = A + (long)(bm + ar) * lda + ao;
  const int br  = tid >> 2;           // n row
  const int bc0 = (tid & 3) << 3;     // k col base
  const __bf16* bp = Bm + (long)(bn + br) * ldb + bo;

  const int nkt   = (K + 31) >> 5;
  const bool aligned = ((ao & 7) == 0) && ((bo & 7) == 0);
  const int nFast = aligned ? (K >> 5) : 0;

  int kt = 0;
  for (; kt < nFast; ++kt) {          // branch-free main loop
    const int k0 = kt << 5;
    *(v8bf*)&As[ar][ac0]     = *(const v8bf*)(ap + k0 + ac0);
    *(v8bf*)&As[ar][ac0 + 8] = *(const v8bf*)(ap + k0 + ac0 + 8);
    *(v8bf*)&Bs[br][bc0]     = *(const v8bf*)(bp + k0 + bc0);
    __syncthreads();
    tile_mma(As, Bs, lane, wm, wn, acc);
    __syncthreads();
  }
  for (; kt < nkt; ++kt) {            // remainder / unaligned: guarded
    const int k0 = kt << 5;
    v8bf o0, o1, ob;
#pragma unroll
    for (int j = 0; j < 8; ++j) {
      const int k = k0 + ac0 + j;
      o0[j] = (k < K) ? ap[k] : (__bf16)0.0f;
    }
#pragma unroll
    for (int j = 0; j < 8; ++j) {
      const int k = k0 + ac0 + 8 + j;
      o1[j] = (k < K) ? ap[k] : (__bf16)0.0f;
    }
#pragma unroll
    for (int j = 0; j < 8; ++j) {
      const int k = k0 + bc0 + j;
      ob[j] = (k < K) ? bp[k] : (__bf16)0.0f;
    }
    *(v8bf*)&As[ar][ac0]     = o0;
    *(v8bf*)&As[ar][ac0 + 8] = o1;
    *(v8bf*)&Bs[br][bc0]     = ob;
    __syncthreads();
    tile_mma(As, Bs, lane, wm, wn, acc);
    __syncthreads();
  }

  const int nl   = lane & 15;
  const int mofs = (lane < 16) ? 0 : 8;
#pragma unroll
  for (int i = 0; i < 2; ++i)
#pragma unroll
    for (int j = 0; j < 2; ++j) {
      const int gn = bn + wn * 32 + j * 16 + nl;
#pragma unroll
      for (int r = 0; r < 8; ++r) {
        const int gm = bm + wm * 32 + i * 16 + mofs + r;
        float v = acc[i][j][r] * scale;
        if (relu) v = fmaxf(v, 0.0f);
        out[(long)gm * ldo + gn] = v;
      }
    }
}

// ---------------------------------------------------------------------------
// gemm_av: out[m, oo+n] += sum_k A[m,k] * B[k, bo+n]   (n < Nc)
//   A fp32 (softmaxed scores, converted to bf16 in LDS), B bf16 (values).
//   K is always a multiple of 32 here (K == NN), so no K-tail is needed,
//   but keep a guarded tail for generality.
// ---------------------------------------------------------------------------
__global__ __launch_bounds__(256) void gemm_av(
    const float* __restrict__ A, int lda,
    const __bf16* __restrict__ Bm, int ldb, int bo,
    float* __restrict__ out, int ldo, int oo,
    int K, int Nc)
{
  __shared__ __align__(16) __bf16 As[128][40];
  __shared__ __align__(16) __bf16 Bs[64][40];   // [n][k]
  const int tid  = threadIdx.x;
  const int lane = tid & 31;
  const int wave = tid >> 5;
  const int wm = wave >> 1, wn = wave & 1;
  const int bm = blockIdx.x * 128;
  const int bn = blockIdx.y * 64;

  v8f acc[2][2];
#pragma unroll
  for (int i = 0; i < 2; ++i)
#pragma unroll
    for (int j = 0; j < 2; ++j)
#pragma unroll
      for (int r = 0; r < 8; ++r) acc[i][j][r] = 0.0f;

  const int ar  = tid >> 1;
  const int ac0 = (tid & 1) << 4;
  const float* ap = A + (long)(bm + ar) * lda;
  const int br  = tid >> 3;
  const int bc0 = (tid & 7) << 3;

  const int nkt   = (K + 31) >> 5;
  const int nFast = K >> 5;
  const bool bFull = (bn + 64 <= Nc) && ((bo & 7) == 0);

  int kt = 0;
  if (bFull) {
    for (; kt < nFast; ++kt) {        // branch-free main loop
      const int k0 = kt << 5;
      const float4* p = (const float4*)(ap + k0 + ac0);
      float4 q0 = p[0], q1 = p[1], q2 = p[2], q3 = p[3];
      *(v8bf*)&As[ar][ac0]     = cvt8(q0, q1);
      *(v8bf*)&As[ar][ac0 + 8] = cvt8(q2, q3);
      v8bf q = *(const v8bf*)(Bm + (long)(k0 + br) * ldb + bo + bn + bc0);
#pragma unroll
      for (int j = 0; j < 8; ++j) Bs[bc0 + j][br] = q[j];
      __syncthreads();
      tile_mma(As, Bs, lane, wm, wn, acc);
      __syncthreads();
    }
  } else {
    for (; kt < nFast; ++kt) {        // n-guarded B, full k
      const int k0 = kt << 5;
      const float4* p = (const float4*)(ap + k0 + ac0);
      float4 q0 = p[0], q1 = p[1], q2 = p[2], q3 = p[3];
      *(v8bf*)&As[ar][ac0]     = cvt8(q0, q1);
      *(v8bf*)&As[ar][ac0 + 8] = cvt8(q2, q3);
      const __bf16* bp = Bm + (long)(k0 + br) * ldb + bo;
#pragma unroll
      for (int j = 0; j < 8; ++j) {
        const int n = bn + bc0 + j;
        Bs[bc0 + j][br] = (n < Nc) ? bp[n] : (__bf16)0.0f;
      }
      __syncthreads();
      tile_mma(As, Bs, lane, wm, wn, acc);
      __syncthreads();
    }
  }
  for (; kt < nkt; ++kt) {            // K remainder: fully guarded
    const int k0 = kt << 5;
    v8bf o0, o1;
#pragma unroll
    for (int j = 0; j < 8; ++j) {
      const int k = k0 + ac0 + j;
      o0[j] = (k < K) ? (__bf16)ap[k] : (__bf16)0.0f;
    }
#pragma unroll
    for (int j = 0; j < 8; ++j) {
      const int k = k0 + ac0 + 8 + j;
      o1[j] = (k < K) ? (__bf16)ap[k] : (__bf16)0.0f;
    }
    *(v8bf*)&As[ar][ac0]     = o0;
    *(v8bf*)&As[ar][ac0 + 8] = o1;
    const int k = k0 + br;
#pragma unroll
    for (int j = 0; j < 8; ++j) {
      const int n = bn + bc0 + j;
      Bs[bc0 + j][br] = (k < K && n < Nc) ? Bm[(long)k * ldb + bo + n]
                                          : (__bf16)0.0f;
    }
    __syncthreads();
    tile_mma(As, Bs, lane, wm, wn, acc);
    __syncthreads();
  }

  const int nl   = lane & 15;
  const int mofs = (lane < 16) ? 0 : 8;
#pragma unroll
  for (int i = 0; i < 2; ++i)
#pragma unroll
    for (int j = 0; j < 2; ++j) {
      const int gn = bn + wn * 32 + j * 16 + nl;
      if (gn < Nc) {
#pragma unroll
        for (int r = 0; r < 8; ++r) {
          const int gm = bm + wm * 32 + i * 16 + mofs + r;
          const long idx = (long)gm * ldo + oo + gn;
          out[idx] += acc[i][j][r];
        }
      }
    }
}

// ---------------------------------------------------------------------------
// softmax_rows: in-place softmax over each row of S [rows, n], n == 4096.
// ---------------------------------------------------------------------------
__global__ __launch_bounds__(256) void softmax_rows(float* __restrict__ S, int n)
{
  __shared__ float red[256];
  const int tid = threadIdx.x;
  float* p = S + (long)blockIdx.x * n;

  float x[16];
  float mx = -1e30f;
#pragma unroll
  for (int i = 0; i < 16; ++i) { x[i] = p[tid + i * 256]; mx = fmaxf(mx, x[i]); }
  red[tid] = mx;
  __syncthreads();
  for (int s = 128; s > 0; s >>= 1) {
    if (tid < s) red[tid] = fmaxf(red[tid], red[tid + s]);
    __syncthreads();
  }
  mx = red[0];
  __syncthreads();

  float sum = 0.0f;
#pragma unroll
  for (int i = 0; i < 16; ++i) { x[i] = __expf(x[i] - mx); sum += x[i]; }
  red[tid] = sum;
  __syncthreads();
  for (int s = 128; s > 0; s >>= 1) {
    if (tid < s) red[tid] += red[tid + s];
    __syncthreads();
  }
  const float inv = 1.0f / red[0];
#pragma unroll
  for (int i = 0; i < 16; ++i) p[tid + i * 256] = x[i] * inv;
}

// ---------------------------------------------------------------------------
// gn_stats: per-feature GraphNorm statistics over `rows` rows of X [rows, ld].
// stats[f*4+0]=a (ms*mean), +1=t (rsqrt(var+eps)), +2=inv L2 norm of the
// normalized column (analytic, only when withNorm).  One block per feature.
// ---------------------------------------------------------------------------
__global__ __launch_bounds__(256) void gn_stats(
    const float* __restrict__ X, int ld, int rows,
    const float* __restrict__ w, const float* __restrict__ b,
    const float* __restrict__ ms, float* __restrict__ stats, int withNorm)
{
  __shared__ float r1[256], r2[256];
  const int f = blockIdx.x;
  const int tid = threadIdx.x;
  float sx = 0.0f, sx2 = 0.0f;
  for (int r = tid; r < rows; r += 256) {
    const float v = X[(long)r * ld + f];
    sx += v; sx2 += v * v;
  }
  r1[tid] = sx; r2[tid] = sx2;
  __syncthreads();
  for (int s = 128; s > 0; s >>= 1) {
    if (tid < s) { r1[tid] += r1[tid + s]; r2[tid] += r2[tid + s]; }
    __syncthreads();
  }
  if (tid == 0) {
    const float Sx = r1[0], Sx2 = r2[0];
    const float mean = Sx / rows;
    const float a = ms[f] * mean;
    const float var = Sx2 / rows - 2.0f * a * mean + a * a;
    const float t = rsqrtf(var + 1e-5f);
    float inv = 1.0f;
    if (withNorm) {
      const float g = t * w[f], bb = b[f];
      const float sumsq = g * g * (var * rows)
                        + 2.0f * g * bb * (Sx - (float)rows * a)
                        + (float)rows * bb * bb;
      inv = rsqrtf(fmaxf(sumsq, 1e-30f));
    }
    stats[f * 4 + 0] = a;
    stats[f * 4 + 1] = t;
    stats[f * 4 + 2] = inv;
  }
}

// gn_apply: in-place GraphNorm over X [rows, ld]; also writes bf16 copy XB.
__global__ __launch_bounds__(256) void gn_apply(
    float* __restrict__ X, __bf16* __restrict__ XB, int ld, int rows, int D,
    const float* __restrict__ w, const float* __restrict__ b,
    const float* __restrict__ stats)
{
  const long idx = (long)blockIdx.x * 256 + threadIdx.x;
  if (idx >= (long)rows * D) return;
  const int r = (int)(idx / D);
  const int f = (int)(idx % D);
  const float a = stats[f * 4], t = stats[f * 4 + 1];
  const long p = (long)r * ld + f;
  const float y = (X[p] - a) * t * w[f] + b[f];
  X[p] = y;
  XB[p] = (__bf16)y;
}

// gn2_transpose: G[r, n] = bf16( gn2(X)[n, r] * inv_norm_r )
__global__ __launch_bounds__(256) void gn2_transpose(
    const float* __restrict__ X, const float* __restrict__ w,
    const float* __restrict__ b, const float* __restrict__ stats,
    __bf16* __restrict__ G)
{
  const long idx = (long)blockIdx.x * 256 + threadIdx.x;
  if (idx >= (long)HRD * NN) return;
  const int r = (int)(idx / NN);
  const int n = (int)(idx % NN);
  const float a = stats[r * 4], t = stats[r * 4 + 1], inv = stats[r * 4 + 2];
  const float y = (X[(long)n * HRD + r] - a) * t * w[r] + b[r];
  G[(long)r * NN + n] = (__bf16)(y * inv);
}

// ---------------------------------------------------------------------------
extern "C" void kernel_launch(void* const* d_in, const int* in_sizes, int n_in,
                              void* d_out, int out_size, void* d_ws, size_t ws_size,
                              hipStream_t stream)
{
  (void)in_sizes; (void)n_in; (void)out_size; (void)ws_size;
  const float* lr_x = (const float*)d_in[0];
  const float* Wq1 = (const float*)d_in[1];  const float* bq1 = (const float*)d_in[2];
  const float* Wk1 = (const float*)d_in[3];  const float* bk1 = (const float*)d_in[4];
  const float* Wv1 = (const float*)d_in[5];  const float* bv1 = (const float*)d_in[6];
  const float* Ws1 = (const float*)d_in[7];  const float* bs1 = (const float*)d_in[8];
  const float* gn1_w = (const float*)d_in[9];
  const float* gn1_b = (const float*)d_in[10];
  const float* gn1_ms = (const float*)d_in[11];
  const float* Wq2 = (const float*)d_in[12]; const float* bq2 = (const float*)d_in[13];
  const float* Wk2 = (const float*)d_in[14]; const float* bk2 = (const float*)d_in[15];
  const float* Wv2 = (const float*)d_in[16]; const float* bv2 = (const float*)d_in[17];
  const float* Ws2 = (const float*)d_in[18]; const float* bs2 = (const float*)d_in[19];
  const float* gn2_w = (const float*)d_in[20];
  const float* gn2_b = (const float*)d_in[21];
  const float* gn2_ms = (const float*)d_in[22];

  // Workspace layout (~180 MB).
  uint8_t* ws = (uint8_t*)d_ws;
  size_t off = 0;
  auto take = [&](size_t bytes) -> uint8_t* {
    uint8_t* p = ws + off;
    off += (bytes + 255) & ~(size_t)255;
    return p;
  };
  __bf16* Xb    = (__bf16*)take((size_t)NN * LRD * 2);   // lr_x^T as bf16
  __bf16* Q1b   = (__bf16*)take((size_t)NN * LD1P * 2);
  __bf16* K1b   = (__bf16*)take((size_t)NN * LD1P * 2);
  __bf16* V1b   = (__bf16*)take((size_t)NN * LD1P * 2);
  float*  S1f   = (float*) take((size_t)NN * LD1P * 4);
  __bf16* H1b   = (__bf16*)take((size_t)NN * LD1P * 2);  // h1 (post-GN) bf16
  float*  scores= (float*) take((size_t)NN * NN * 4);
  __bf16* Q2b   = (__bf16*)take((size_t)NN * HRD * 2);
  __bf16* K2b   = (__bf16*)take((size_t)NN * HRD * 2);
  __bf16* V2b   = (__bf16*)take((size_t)NN * HRD * 2);
  float*  S2f   = (float*) take((size_t)NN * HRD * 4);
  float*  st1   = (float*) take((size_t)DD1 * 4 * 4);
  float*  st2   = (float*) take((size_t)HRD * 4 * 4);
  __bf16* Gbf   = (__bf16*)take((size_t)HRD * NN * 2);

  const float scale1 = 1.0f / sqrtf((float)CH1);
  const float scale2 = 1.0f / sqrtf((float)CH2);

  // ---- Transpose lr_x [LRD, NN] -> Xb bf16 [NN, LRD] (one-time, tiny) ----
  transpose_f32_to_bf16<<<dim3(NN / 32, LRD / 32), 256, 0, stream>>>(lr_x, Xb, LRD, NN);

  // ---- Layer 1 linears ----
  dim3 gLin1(NN / 128, LD1P / 64);
  gemm_lin<<<gLin1, 256, 0, stream>>>(Xb, LRD, Wq1, bq1, nullptr, Q1b, LD1P, LRD, DD1);
  gemm_lin<<<gLin1, 256, 0, stream>>>(Xb, LRD, Wk1, bk1, nullptr, K1b, LD1P, LRD, DD1);
  gemm_lin<<<gLin1, 256, 0, stream>>>(Xb, LRD, Wv1, bv1, nullptr, V1b, LD1P, LRD, DD1);
  gemm_lin<<<gLin1, 256, 0, stream>>>(Xb, LRD, Ws1, bs1, S1f, nullptr, LD1P, LRD, DD1);

  // ---- Layer 1 attention (per head, materialized scores) ----
  dim3 gScore(NN / 128, NN / 64);
  for (int h = 0; h < NHEAD; ++h) {
    const int ho = h * CH1;
    gemm_gram<<<gScore, 256, 0, stream>>>(Q1b, LD1P, ho, K1b, LD1P, ho,
                                          scores, NN, CH1, scale1, 0);
    softmax_rows<<<NN, 256, 0, stream>>>(scores, NN);
    gemm_av<<<dim3(NN / 128, 1), 256, 0, stream>>>(scores, NN, V1b, LD1P, ho,
                                                   S1f, LD1P, ho, NN, CH1);
  }

  // ---- GraphNorm 1 (in place on S1f; bf16 copy into H1b) ----
  gn_stats<<<DD1, 256, 0, stream>>>(S1f, LD1P, NN, gn1_w, gn1_b, gn1_ms, st1, 0);
  gn_apply<<<(int)(((long)NN * DD1 + 255) / 256), 256, 0, stream>>>(
      S1f, H1b, LD1P, NN, DD1, gn1_w, gn1_b, st1);

  // ---- Layer 2 linears (A = h1 bf16, K = 200) ----
  dim3 gLin2(NN / 128, HRD / 64);
  gemm_lin<<<gLin2, 256, 0, stream>>>(H1b, LD1P, Wq2, bq2, nullptr, Q2b, HRD, DD1, HRD);
  gemm_lin<<<gLin2, 256, 0, stream>>>(H1b, LD1P, Wk2, bk2, nullptr, K2b, HRD, DD1, HRD);
  gemm_lin<<<gLin2, 256, 0, stream>>>(H1b, LD1P, Wv2, bv2, nullptr, V2b, HRD, DD1, HRD);
  gemm_lin<<<gLin2, 256, 0, stream>>>(H1b, LD1P, Ws2, bs2, S2f, nullptr, HRD, DD1, HRD);

  // ---- Layer 2 attention (per head, d = 512) ----
  for (int h = 0; h < NHEAD; ++h) {
    const int ho = h * CH2;
    gemm_gram<<<gScore, 256, 0, stream>>>(Q2b, HRD, ho, K2b, HRD, ho,
                                          scores, NN, CH2, scale2, 0);
    softmax_rows<<<NN, 256, 0, stream>>>(scores, NN);
    gemm_av<<<dim3(NN / 128, CH2 / 64), 256, 0, stream>>>(scores, NN, V2b, HRD, ho,
                                                          S2f, HRD, ho, NN, CH2);
  }

  // ---- GraphNorm 2 + transpose + row L2-normalize -> Gbf [HRD, NN] bf16 ----
  gn_stats<<<HRD, 256, 0, stream>>>(S2f, HRD, NN, gn2_w, gn2_b, gn2_ms, st2, 1);
  gn2_transpose<<<(int)(((long)HRD * NN + 255) / 256), 256, 0, stream>>>(
      S2f, gn2_w, gn2_b, st2, Gbf);

  // ---- Output: relu(G @ G^T)  [2048, 2048] fp32 ----
  gemm_gram<<<dim3(HRD / 128, HRD / 64), 256, 0, stream>>>(
      Gbf, NN, 0, Gbf, NN, 0, (float*)d_out, HRD, NN, 1.0f, 1);
}
